// GreedySearchDecoder_50543175139657
// MI455X (gfx1250) — compile-verified, matrix operand order checked by reading, not run on hardware
//
#include <hip/hip_runtime.h>
#include <hip/hip_bf16.h>

// ---------------------------------------------------------------------------
// GreedySearchDecoder for MI455X (gfx1250, wave32, WMMA).
//  - Encoder input projections (256x1500x{512,1024}) use v_wmma_f32_16x16x32_f16
//    with the A-slab staged in LDS via CDNA5 async-to-LDS loads
//  - GRU recurrences / GEMVs use wave-per-row fp32 dot products (wave32 shfl)
//  - Vocab projection reads f16 out_W (51MB, L2-resident on 192MB L2)
// ---------------------------------------------------------------------------

#define V_VOCAB 50257
#define HH      500
#define H3      1500
#define SS      256
#define MAXLEN  64
#define KP0     512     // padded K for H=500
#define KP1     1024    // padded K for 2H=1000
#define NPAD    1504    // padded 3H=1500 (94 tiles of 16)
#define GEMM_NW 8       // waves per GEMM block

typedef _Float16 v8h  __attribute__((ext_vector_type(8)));
typedef _Float16 v16h __attribute__((ext_vector_type(16)));
typedef float    v8f  __attribute__((ext_vector_type(8)));
typedef __attribute__((address_space(3))) _Float16 lds_f16;

// ---------------- helpers ----------------
__device__ __forceinline__ float wred_sum(float v) {
    for (int o = 16; o > 0; o >>= 1) v += __shfl_down(v, o, 32);
    return v;
}
__device__ __forceinline__ float sigmoidf_(float x) {
    return 1.0f / (1.0f + __expf(-x));
}

// ---------------- conversion / layout kernels ----------------
__global__ void cvt_pad_f16(const float* __restrict__ src, _Float16* __restrict__ dst,
                            int R, int C, int Rp, int Cp) {
    size_t i = (size_t)blockIdx.x * blockDim.x + threadIdx.x;
    size_t n = (size_t)Rp * Cp;
    if (i >= n) return;
    int r = (int)(i / Cp), c = (int)(i % Cp);
    dst[i] = (r < R && c < C) ? (_Float16)src[(size_t)r * C + c] : (_Float16)0.0f;
}

__global__ void embed_gather_f16(const int* __restrict__ seq, const float* __restrict__ emb,
                                 _Float16* __restrict__ Xh) {
    size_t i = (size_t)blockIdx.x * blockDim.x + threadIdx.x;
    if (i >= (size_t)SS * KP0) return;
    int s = (int)(i >> 9), k = (int)(i & 511);
    int tok = seq[s];
    Xh[i] = (k < HH) ? (_Float16)emb[(size_t)tok * HH + k] : (_Float16)0.0f;
}

__global__ void x1_build_f16(const float* __restrict__ y0f, const float* __restrict__ y0b,
                             _Float16* __restrict__ x1h) {
    size_t i = (size_t)blockIdx.x * blockDim.x + threadIdx.x;
    if (i >= (size_t)SS * KP1) return;
    int s = (int)(i >> 10), k = (int)(i & 1023);
    float v = 0.0f;
    if (k < HH)            v = y0f[(size_t)s * HH + k];
    else if (k < 2 * HH)   v = y0b[(size_t)s * HH + (k - HH)];
    x1h[i] = (_Float16)v;
}

__global__ void add_vec(const float* __restrict__ a, const float* __restrict__ b,
                        float* __restrict__ o, int n) {
    int i = blockIdx.x * blockDim.x + threadIdx.x;
    if (i < n) o[i] = a[i] + b[i];
}

__global__ void init_slots(unsigned long long* __restrict__ smax, float* __restrict__ ssum) {
    int i = threadIdx.x;
    if (i < MAXLEN) { smax[i] = 0ull; ssum[i] = 0.0f; }
}

__global__ void init_dec(const float* __restrict__ h0fT, const float* __restrict__ h0bT,
                         float* __restrict__ hA, int* __restrict__ d_tok) {
    int i = blockIdx.x * blockDim.x + threadIdx.x;
    if (i < HH) { hA[i] = h0fT[i]; hA[HH + i] = h0bT[i]; }
    if (i == 0) *d_tok = 1;  // SOS
}

// ---------------- WMMA GEMM:  C[M x N] = Xh[M x Kp] * Wh[N x Kp]^T + bias[N] ----------------
// Block = 8 waves sharing one m-tile. The 16-row A slab (contiguous, 16*Kp halfs)
// is staged into LDS once per block with global_load_async_to_lds_b128 (ASYNCcnt),
// then each wave computes one 16x16 tile per 8-wide n-tile group, reading A
// fragments from LDS (ds_load_b128) and streaming B from L2.
// Fragment layout per CDNA5 ISA (16-bit A 16x32): lane<16 -> K {0..7,16..23},
// lane>=16 -> K {8..15,24..31}; B mirrors A with row index = output column.
__global__ __launch_bounds__(256)
void wmma_gemm_bias(const _Float16* __restrict__ X, const _Float16* __restrict__ W,
                    const float* __restrict__ bias, float* __restrict__ C,
                    int M, int Kp, int N) {
    __shared__ _Float16 Alds[16 * KP1];          // 32KB max slab
    const int lane = threadIdx.x & 31;
    const int wv   = threadIdx.x >> 5;
    const int mtiles = M >> 4;
    const int ntiles = (N + 15) >> 4;
    const int mt = blockIdx.x % mtiles;
    const int nt = (blockIdx.x / mtiles) * GEMM_NW + wv;

    // --- cooperative async fill of the A slab ---
    {
        const unsigned lds0 =
            (unsigned)(size_t)(lds_f16*)Alds;     // raw LDS byte offset (AS3 ptrtoint)
        const unsigned long long g0 =
            (unsigned long long)(const void*)(X + (size_t)mt * 16 * Kp);
        for (int c = threadIdx.x; c < 2 * Kp; c += 256) {     // 16B chunks
            unsigned b = (unsigned)c * 16u;
            asm volatile("global_load_async_to_lds_b128 %0, %1, %2"
                         :: "v"(lds0 + b), "v"(b), "s"(g0) : "memory");
        }
        asm volatile("s_wait_asynccnt 0x0" ::: "memory");
        __syncthreads();
    }

    if (nt < ntiles) {
        const int hs  = lane >> 4;       // half-wave select
        const int l15 = lane & 15;
        const _Float16* pa = Alds + (size_t)l15 * Kp + hs * 8;               // LDS
        const _Float16* pb = W + (size_t)(nt * 16 + l15) * Kp + hs * 8;      // global

        v8f acc = {};
        for (int k0 = 0; k0 < Kp; k0 += 32) {
            v8h a0 = *(const v8h*)(pa + k0);
            v8h a1 = *(const v8h*)(pa + k0 + 16);
            v8h b0 = *(const v8h*)(pb + k0);
            v8h b1 = *(const v8h*)(pb + k0 + 16);
            v16h A, B;
#pragma unroll
            for (int i = 0; i < 8; ++i) { A[i] = a0[i]; A[8 + i] = a1[i]; B[i] = b0[i]; B[8 + i] = b1[i]; }
            acc = __builtin_amdgcn_wmma_f32_16x16x32_f16(false, A, false, B, (short)0, acc, false, false);
        }
        const int ncol = nt * 16 + l15;
        if (ncol < N) {
            const float bv = bias[ncol];
#pragma unroll
            for (int r = 0; r < 8; ++r) {
                int mrow = mt * 16 + r + hs * 8;   // D layout: vgpr r -> M=r (+8 upper half-wave)
                C[(size_t)mrow * N + ncol] = acc[r] + bv;
            }
        }
    }
}

// ---------------- GRU scan (one workgroup per direction, sequential over S steps) ----------------
__global__ __launch_bounds__(1024)
void gru_scan(const float* __restrict__ gx_f, const float* __restrict__ gx_b,
              const _Float16* __restrict__ Whh_f, const _Float16* __restrict__ Whh_b,
              const float* __restrict__ bhh_f, const float* __restrict__ bhh_b,
              float* __restrict__ y_f, float* __restrict__ y_b,
              float* __restrict__ hT_f, float* __restrict__ hT_b) {
    const int dir = blockIdx.x;
    const float*    gx  = dir ? gx_b  : gx_f;
    const _Float16* Whh = dir ? Whh_b : Whh_f;
    const float*    bhh = dir ? bhh_b : bhh_f;
    float*          y   = dir ? y_b   : y_f;
    float*          hT  = dir ? hT_b  : hT_f;

    __shared__ float h[KP0];         // zero-padded 500..511 stay 0 forever
    __shared__ float gh[NPAD];
    const int tid = threadIdx.x;
    if (tid < KP0) h[tid] = 0.0f;
    __syncthreads();

    for (int s = 0; s < SS; ++s) {
        const int t = dir ? (SS - 1 - s) : s;
        // prefetch next step's gate row (6KB = 47 cachelines) into caches
        {
            const int tn = dir ? (t > 0 ? t - 1 : 0) : (t + 1 < SS ? t + 1 : t);
            const char* pf = (const char*)(gx + (size_t)tn * H3);
            if (tid < 47) __builtin_prefetch(pf + tid * 128, 0, 0);
        }
        // gh = Whh @ h + bhh   (f16 weights via v8h, fp32 accumulate, h from LDS)
        for (int r = tid; r < H3; r += 1024) {
            const v8h* w8 = (const v8h*)(Whh + (size_t)r * KP0);
            float acc = 0.0f;
#pragma unroll 4
            for (int kc = 0; kc < KP0 / 8; ++kc) {
                v8h wvv = w8[kc];
                float4 h0 = *(const float4*)&h[kc * 8];
                float4 h1 = *(const float4*)&h[kc * 8 + 4];
                acc = fmaf((float)wvv[0], h0.x, acc);
                acc = fmaf((float)wvv[1], h0.y, acc);
                acc = fmaf((float)wvv[2], h0.z, acc);
                acc = fmaf((float)wvv[3], h0.w, acc);
                acc = fmaf((float)wvv[4], h1.x, acc);
                acc = fmaf((float)wvv[5], h1.y, acc);
                acc = fmaf((float)wvv[6], h1.z, acc);
                acc = fmaf((float)wvv[7], h1.w, acc);
            }
            gh[r] = acc + bhh[r];
        }
        __syncthreads();
        // gates (PyTorch order r,z,n)
        if (tid < HH) {
            const float* g = gx + (size_t)t * H3;
            float rg = sigmoidf_(g[tid]          + gh[tid]);
            float zg = sigmoidf_(g[tid + HH]     + gh[tid + HH]);
            float ng = tanhf   (g[tid + 2 * HH] + rg * gh[tid + 2 * HH]);
            float h2 = (1.0f - zg) * ng + zg * h[tid];
            y[(size_t)t * HH + tid] = h2;
            h[tid] = h2;
        }
        __syncthreads();
    }
    if (tid < HH) hT[tid] = h[tid];
}

// ---------------- decoder GRU cell (wave-per-output-j) ----------------
__global__ __launch_bounds__(128)
void dec_cell(const float* __restrict__ Wih, const float* __restrict__ Whh,
              const float* __restrict__ bih, const float* __restrict__ bhh,
              const float* __restrict__ xvec, int use_emb,
              const float* __restrict__ emb, const int* __restrict__ tokp,
              const float* __restrict__ h_in, float* __restrict__ h_out) {
    const int wid  = (blockIdx.x * blockDim.x + threadIdx.x) >> 5;
    const int lane = threadIdx.x & 31;
    if (wid >= HH) return;
    const float* x = use_emb ? (emb + (size_t)(*tokp) * HH) : xvec;

    float ar = 0, az = 0, an = 0, br = 0, bz = 0, bn = 0;
    for (int k = lane; k < HH; k += 32) {
        float xv = x[k], hv = h_in[k];
        ar += Wih[(size_t)(wid)            * HH + k] * xv;
        az += Wih[(size_t)(wid + HH)       * HH + k] * xv;
        an += Wih[(size_t)(wid + 2 * HH)   * HH + k] * xv;
        br += Whh[(size_t)(wid)            * HH + k] * hv;
        bz += Whh[(size_t)(wid + HH)       * HH + k] * hv;
        bn += Whh[(size_t)(wid + 2 * HH)   * HH + k] * hv;
    }
    ar = wred_sum(ar); az = wred_sum(az); an = wred_sum(an);
    br = wred_sum(br); bz = wred_sum(bz); bn = wred_sum(bn);
    if (lane == 0) {
        float r = sigmoidf_(ar + bih[wid]          + br + bhh[wid]);
        float z = sigmoidf_(az + bih[wid + HH]     + bz + bhh[wid + HH]);
        float n = tanhf   (an + bih[wid + 2 * HH] + r * (bn + bhh[wid + 2 * HH]));
        h_out[wid] = (1.0f - z) * n + z * h_in[wid];
    }
}

// ---------------- attention: scores -> softmax -> ctx (single block) ----------------
__global__ __launch_bounds__(1024)
void attn_kernel(const float* __restrict__ enc_out, const float* __restrict__ rnn,
                 float* __restrict__ ctx) {
    __shared__ float w[SS];
    __shared__ float red[32];
    __shared__ float sMax, sSum;
    const int tid = threadIdx.x, wid = tid >> 5, lane = tid & 31;

    for (int i = wid; i < SS; i += 32) {
        float acc = 0.0f;
        for (int k = lane; k < HH; k += 32) acc += enc_out[(size_t)i * HH + k] * rnn[k];
        acc = wred_sum(acc);
        if (lane == 0) w[i] = acc;
    }
    __syncthreads();
    // max
    float sv = (tid < SS) ? w[tid] : -3.4e38f;
    for (int o = 16; o > 0; o >>= 1) sv = fmaxf(sv, __shfl_down(sv, o, 32));
    if (lane == 0) red[wid] = sv;
    __syncthreads();
    if (tid == 0) { float m = red[0]; for (int i = 1; i < 8; ++i) m = fmaxf(m, red[i]); sMax = m; }
    __syncthreads();
    // sum of exp
    float e = (tid < SS) ? __expf(w[tid] - sMax) : 0.0f;
    float es = wred_sum(e);
    if (lane == 0) red[wid] = es;
    __syncthreads();
    if (tid == 0) { float s = 0; for (int i = 0; i < 8; ++i) s += red[i]; sSum = s; }
    __syncthreads();
    if (tid < SS) w[tid] = e / sSum;
    __syncthreads();
    // ctx = w @ enc_out
    for (int k = tid; k < HH; k += 1024) {
        float acc = 0.0f;
        for (int i = 0; i < SS; ++i) acc += w[i] * enc_out[(size_t)i * HH + k];
        ctx[k] = acc;
    }
}

// ---------------- concat layer: c = tanh(Wc @ [rnn; ctx] + bc), padded to 512 ----------------
__global__ __launch_bounds__(128)
void concat_tanh(const float* __restrict__ Wc, const float* __restrict__ bc,
                 const float* __restrict__ rnn, const float* __restrict__ ctx,
                 float* __restrict__ c) {
    const int wid  = (blockIdx.x * blockDim.x + threadIdx.x) >> 5;
    const int lane = threadIdx.x & 31;
    if (wid >= KP0) return;
    if (wid >= HH) { if (lane == 0) c[wid] = 0.0f; return; }
    float acc = 0.0f;
    for (int k = lane; k < HH; k += 32) {
        acc += Wc[(size_t)wid * (2 * HH) + k]      * rnn[k];
        acc += Wc[(size_t)wid * (2 * HH) + HH + k] * ctx[k];
    }
    acc = wred_sum(acc);
    if (lane == 0) c[wid] = tanhf(acc + bc[wid]);
}

// ---------------- vocab projection + packed argmax (wave-per-row over f16 out_W) ----------------
__global__ __launch_bounds__(256)
void logits_argmax(const _Float16* __restrict__ Wo, const float* __restrict__ ob,
                   const float* __restrict__ c, float* __restrict__ logits,
                   unsigned long long* __restrict__ slot) {
    const int wid  = (blockIdx.x * blockDim.x + threadIdx.x) >> 5;
    const int lane = threadIdx.x & 31;
    if (wid >= V_VOCAB) return;
    const _Float16* w = Wo + (size_t)wid * KP0 + lane * 16;   // 16 contiguous halfs per lane
    float acc = 0.0f;
    v8h w0 = *(const v8h*)(w);
    v8h w1 = *(const v8h*)(w + 8);
#pragma unroll
    for (int i = 0; i < 8; ++i) {
        acc += (float)w0[i] * c[lane * 16 + i];
        acc += (float)w1[i] * c[lane * 16 + 8 + i];
    }
    acc = wred_sum(acc);
    if (lane == 0) {
        float lg = acc + ob[wid];
        logits[wid] = lg;
        unsigned u = __float_as_uint(lg);
        u = (u & 0x80000000u) ? ~u : (u | 0x80000000u);            // order-preserving map
        unsigned long long key = ((unsigned long long)u << 32) |
                                 (unsigned long long)(0xFFFFFFFFu - (unsigned)wid);  // ties -> lowest idx
        atomicMax(slot, key);
    }
}

__global__ __launch_bounds__(256)
void exp_sum(const float* __restrict__ logits, const unsigned long long* __restrict__ slot,
             float* __restrict__ probs, float* __restrict__ sum) {
    __shared__ float r8[8];
    const int i = blockIdx.x * blockDim.x + threadIdx.x;
    const int wid = threadIdx.x >> 5, lane = threadIdx.x & 31;
    unsigned u = (unsigned)((*slot) >> 32);
    unsigned fu = (u & 0x80000000u) ? (u & 0x7FFFFFFFu) : ~u;
    const float mx = __uint_as_float(fu);
    float e = 0.0f;
    if (i < V_VOCAB) { e = __expf(logits[i] - mx); probs[i] = e; }
    float b = wred_sum(e);
    if (lane == 0) r8[wid] = b;
    __syncthreads();
    if (threadIdx.x == 0) {
        float s = 0; for (int k = 0; k < 8; ++k) s += r8[k];
        atomicAdd(sum, s);
    }
}

__global__ __launch_bounds__(256)
void norm_tok(float* __restrict__ probs, const float* __restrict__ sum,
              const unsigned long long* __restrict__ slot,
              float* __restrict__ tok_out, int* __restrict__ d_tok) {
    const int i = blockIdx.x * blockDim.x + threadIdx.x;
    if (i < V_VOCAB) probs[i] *= (1.0f / *sum);
    if (blockIdx.x == 0 && threadIdx.x == 0) {
        unsigned idx = 0xFFFFFFFFu - (unsigned)((*slot) & 0xFFFFFFFFull);
        *d_tok = (int)idx;
        *tok_out = (float)idx;
    }
}

// ---------------------------------------------------------------------------
extern "C" void kernel_launch(void* const* d_in, const int* in_sizes, int n_in,
                              void* d_out, int out_size, void* d_ws, size_t ws_size,
                              hipStream_t stream) {
    (void)in_sizes; (void)n_in; (void)out_size; (void)ws_size;

    const int*   seq      = (const int*)d_in[0];
    const float* emb      = (const float*)d_in[3];
    const float* e0f_Wih  = (const float*)d_in[4];
    const float* e0f_Whh  = (const float*)d_in[5];
    const float* e0f_bih  = (const float*)d_in[6];
    const float* e0f_bhh  = (const float*)d_in[7];
    const float* e0b_Wih  = (const float*)d_in[8];
    const float* e0b_Whh  = (const float*)d_in[9];
    const float* e0b_bih  = (const float*)d_in[10];
    const float* e0b_bhh  = (const float*)d_in[11];
    const float* e1f_Wih  = (const float*)d_in[12];
    const float* e1f_Whh  = (const float*)d_in[13];
    const float* e1f_bih  = (const float*)d_in[14];
    const float* e1f_bhh  = (const float*)d_in[15];
    const float* e1b_Wih  = (const float*)d_in[16];
    const float* e1b_Whh  = (const float*)d_in[17];
    const float* e1b_bih  = (const float*)d_in[18];
    const float* e1b_bhh  = (const float*)d_in[19];
    const float* dWih     = (const float*)d_in[20];  // (2,1500,500)
    const float* dWhh     = (const float*)d_in[21];
    const float* dbih     = (const float*)d_in[22];  // (2,1500)
    const float* dbhh     = (const float*)d_in[23];
    const float* cW       = (const float*)d_in[24];  // (500,1000)
    const float* cb       = (const float*)d_in[25];
    const float* oW       = (const float*)d_in[26];  // (V,500)
    const float* ob       = (const float*)d_in[27];

    float* out = (float*)d_out;           // [0..63] tokens, then 64*V probs

    // --- workspace sub-allocation (256B aligned) ---
    size_t off = 0;
    char* base = (char*)d_ws;
    auto alloc = [&](size_t bytes) -> void* {
        void* p = base + off;
        off += (bytes + 255) & ~(size_t)255;
        return p;
    };
    _Float16* exh  = (_Float16*)alloc((size_t)SS * KP0 * 2);
    _Float16* w0f  = (_Float16*)alloc((size_t)NPAD * KP0 * 2);
    _Float16* w0b  = (_Float16*)alloc((size_t)NPAD * KP0 * 2);
    _Float16* w1f  = (_Float16*)alloc((size_t)NPAD * KP1 * 2);
    _Float16* w1b  = (_Float16*)alloc((size_t)NPAD * KP1 * 2);
    _Float16* u0f  = (_Float16*)alloc((size_t)H3 * KP0 * 2);
    _Float16* u0b  = (_Float16*)alloc((size_t)H3 * KP0 * 2);
    _Float16* u1f  = (_Float16*)alloc((size_t)H3 * KP0 * 2);
    _Float16* u1b  = (_Float16*)alloc((size_t)H3 * KP0 * 2);
    _Float16* oWh  = (_Float16*)alloc((size_t)V_VOCAB * KP0 * 2);
    float* gx0f    = (float*)alloc((size_t)SS * H3 * 4);
    float* gx0b    = (float*)alloc((size_t)SS * H3 * 4);
    float* gx1f    = (float*)alloc((size_t)SS * H3 * 4);
    float* gx1b    = (float*)alloc((size_t)SS * H3 * 4);
    float* y0f     = (float*)alloc((size_t)SS * HH * 4);
    float* y0b     = (float*)alloc((size_t)SS * HH * 4);
    float* y1f     = (float*)alloc((size_t)SS * HH * 4);
    float* y1b     = (float*)alloc((size_t)SS * HH * 4);
    _Float16* x1h  = (_Float16*)alloc((size_t)SS * KP1 * 2);
    float* enc_out = (float*)alloc((size_t)SS * HH * 4);
    float* h0fT    = (float*)alloc(HH * 4);
    float* h0bT    = (float*)alloc(HH * 4);
    float* dumT    = (float*)alloc(2 * KP0 * 4);
    float* hA      = (float*)alloc(2 * HH * 4);
    float* hB      = (float*)alloc(2 * HH * 4);
    float* ctx     = (float*)alloc(KP0 * 4);
    float* cvec    = (float*)alloc(KP0 * 4);
    float* lbuf    = (float*)alloc((size_t)V_VOCAB * 4);
    unsigned long long* smax = (unsigned long long*)alloc(MAXLEN * 8);
    float* ssum    = (float*)alloc(MAXLEN * 4);
    int* d_tok     = (int*)alloc(256);

    auto grid1d = [](size_t n, int bs) { return (unsigned)((n + bs - 1) / bs); };

    // --- per-launch init (graph-replay safe) ---
    init_slots<<<1, 64, 0, stream>>>(smax, ssum);

    // --- weight conversions to f16 (padded) ---
    cvt_pad_f16<<<grid1d((size_t)NPAD * KP0, 256), 256, 0, stream>>>(e0f_Wih, w0f, H3, HH, NPAD, KP0);
    cvt_pad_f16<<<grid1d((size_t)NPAD * KP0, 256), 256, 0, stream>>>(e0b_Wih, w0b, H3, HH, NPAD, KP0);
    cvt_pad_f16<<<grid1d((size_t)NPAD * KP1, 256), 256, 0, stream>>>(e1f_Wih, w1f, H3, 2 * HH, NPAD, KP1);
    cvt_pad_f16<<<grid1d((size_t)NPAD * KP1, 256), 256, 0, stream>>>(e1b_Wih, w1b, H3, 2 * HH, NPAD, KP1);
    cvt_pad_f16<<<grid1d((size_t)H3 * KP0, 256), 256, 0, stream>>>(e0f_Whh, u0f, H3, HH, H3, KP0);
    cvt_pad_f16<<<grid1d((size_t)H3 * KP0, 256), 256, 0, stream>>>(e0b_Whh, u0b, H3, HH, H3, KP0);
    cvt_pad_f16<<<grid1d((size_t)H3 * KP0, 256), 256, 0, stream>>>(e1f_Whh, u1f, H3, HH, H3, KP0);
    cvt_pad_f16<<<grid1d((size_t)H3 * KP0, 256), 256, 0, stream>>>(e1b_Whh, u1b, H3, HH, H3, KP0);
    cvt_pad_f16<<<grid1d((size_t)V_VOCAB * KP0, 256), 256, 0, stream>>>(oW, oWh, V_VOCAB, HH, V_VOCAB, KP0);

    // --- encoder ---
    embed_gather_f16<<<grid1d((size_t)SS * KP0, 256), 256, 0, stream>>>(seq, emb, exh);

    const int mtiles = SS / 16;
    const int ntiles = (H3 + 15) / 16;                       // 94
    const int ntgroups = (ntiles + GEMM_NW - 1) / GEMM_NW;   // 12
    const unsigned gemm_blocks = (unsigned)(mtiles * ntgroups);
    wmma_gemm_bias<<<gemm_blocks, 256, 0, stream>>>(exh, w0f, e0f_bih, gx0f, SS, KP0, H3);
    wmma_gemm_bias<<<gemm_blocks, 256, 0, stream>>>(exh, w0b, e0b_bih, gx0b, SS, KP0, H3);

    gru_scan<<<2, 1024, 0, stream>>>(gx0f, gx0b, u0f, u0b, e0f_bhh, e0b_bhh,
                                     y0f, y0b, h0fT, h0bT);

    x1_build_f16<<<grid1d((size_t)SS * KP1, 256), 256, 0, stream>>>(y0f, y0b, x1h);
    wmma_gemm_bias<<<gemm_blocks, 256, 0, stream>>>(x1h, w1f, e1f_bih, gx1f, SS, KP1, H3);
    wmma_gemm_bias<<<gemm_blocks, 256, 0, stream>>>(x1h, w1b, e1b_bih, gx1b, SS, KP1, H3);

    gru_scan<<<2, 1024, 0, stream>>>(gx1f, gx1b, u1f, u1b, e1f_bhh, e1b_bhh,
                                     y1f, y1b, dumT, dumT + KP0);

    add_vec<<<grid1d((size_t)SS * HH, 256), 256, 0, stream>>>(y1f, y1b, enc_out, SS * HH);

    // --- decoder (64 greedy steps; token fed back via device memory) ---
    init_dec<<<2, 512, 0, stream>>>(h0fT, h0bT, hA, d_tok);

    float* hcur = hA;
    float* hnxt = hB;
    const unsigned vblocks = grid1d(V_VOCAB, 256);
    for (int t = 0; t < MAXLEN; ++t) {
        // layer 0 cell: x = emb[tok]
        dec_cell<<<grid1d((size_t)HH * 32, 128), 128, 0, stream>>>(
            dWih, dWhh, dbih, dbhh, (const float*)nullptr, 1, emb, d_tok, hcur, hnxt);
        // layer 1 cell: x = new h0
        dec_cell<<<grid1d((size_t)HH * 32, 128), 128, 0, stream>>>(
            dWih + (size_t)H3 * HH, dWhh + (size_t)H3 * HH, dbih + H3, dbhh + H3,
            hnxt, 0, emb, d_tok, hcur + HH, hnxt + HH);

        const float* rnn = hnxt + HH;
        attn_kernel<<<1, 1024, 0, stream>>>(enc_out, rnn, ctx);
        concat_tanh<<<grid1d((size_t)KP0 * 32, 128), 128, 0, stream>>>(cW, cb, rnn, ctx, cvec);

        float* probs_t = out + MAXLEN + (size_t)t * V_VOCAB;
        logits_argmax<<<grid1d((size_t)V_VOCAB * 32, 256), 256, 0, stream>>>(
            oWh, ob, cvec, lbuf, smax + t);
        exp_sum<<<vblocks, 256, 0, stream>>>(lbuf, smax + t, probs_t, ssum + t);
        norm_tok<<<vblocks, 256, 0, stream>>>(probs_t, ssum + t, smax + t, out + t, d_tok);

        float* tmp = hcur; hcur = hnxt; hnxt = tmp;
    }
}
